// DeformableRead_63788854280195
// MI455X (gfx1250) — matrix operand order repaired and checked by reading, not compiled, not run on hardware
//
#include <hip/hip_runtime.h>
#include <hip/hip_bf16.h>
#include <math.h>

// ---------------- types ----------------
typedef __bf16 bf16;
typedef __bf16 v16bf __attribute__((ext_vector_type(16)));
typedef __bf16 v8bf  __attribute__((ext_vector_type(8)));
typedef float  v8f   __attribute__((ext_vector_type(8)));

// ---------------- problem constants ----------------
#define D_MODEL 192
#define NHEAD   6
#define DH      32
#define BB      2
#define KK      256
#define RR      32
#define UIN     416        // 2*D_MODEL + 32 (fourier PE)
#define G2COLS  224        // 144 offsets + 72 logits, padded to 14*16

// weight tiling (WMMA 16x16x32 bf16)
#define WU_NT 12
#define WU_KT 13           // 416 / 32
#define WDA_NT 14          // 224 / 16
#define WDA_KT 6           // 192 / 32
#define WO_NT 12
#define WO_KT 6
#define FRAG_ELEMS 512     // 32 lanes * 16 bf16
#define WU_ELEMS  (WU_NT*WU_KT*FRAG_ELEMS)    // 79872
#define WDA_ELEMS (WDA_NT*WDA_KT*FRAG_ELEMS)  // 43008
#define WO_ELEMS  (WO_NT*WO_KT*FRAG_ELEMS)    // 36864
#define WS_WDA_OFF (WU_ELEMS)
#define WS_WO_OFF  (WU_ELEMS + WDA_ELEMS)
#define WS_TOTAL   (WU_ELEMS + WDA_ELEMS + WO_ELEMS)  // 159744 bf16 = 312KB

// ---------------- WMMA helpers ----------------
__device__ __forceinline__ v8f wmma_bf16(v16bf a, v16bf b, v8f c) {
  return __builtin_amdgcn_wmma_f32_16x16x32_bf16(
      /*neg_a=*/false, a, /*neg_b=*/false, b,
      /*c_mod=*/(short)0, c, /*reuse_a=*/false, /*reuse_b=*/false);
}

// A fragment (16x32 bf16, row-major in LDS): lane holds row=lane&15, half=lane>>4.
// Elements 0..7  = K [kBase + half*8, +8)
// Elements 8..15 = K [kBase + 16 + half*8, +8)   -> two 16B contiguous runs.
__device__ __forceinline__ v16bf load_a_frag(const bf16* base, int strideElems,
                                             int kBase, int lane) {
  int row  = lane & 15;
  int half = lane >> 4;
  const bf16* p0 = base + row * strideElems + kBase + half * 8;
  v8bf lo = *(const v8bf*)p0;
  v8bf hi = *(const v8bf*)(p0 + 16);
  v16bf a;
#pragma unroll
  for (int i = 0; i < 8; ++i) { a[i] = lo[i]; a[8 + i] = hi[i]; }
  return a;
}

// B fragment pre-swizzled in workspace: lane reads 16 consecutive bf16 (32B).
__device__ __forceinline__ v16bf load_b_frag(const bf16* frag, int lane) {
  const bf16* p = frag + lane * 16;
  v8bf lo = *(const v8bf*)p;
  v8bf hi = *(const v8bf*)(p + 8);
  v16bf b;
#pragma unroll
  for (int i = 0; i < 8; ++i) { b[i] = lo[i]; b[8 + i] = hi[i]; }
  return b;
}

__device__ __forceinline__ float bilin(const float* fmap, int HW, float px, float py) {
  float x0f = floorf(px), y0f = floorf(py);
  float wx1 = px - x0f,  wy1 = py - y0f;
  float wx0 = 1.0f - wx1, wy0 = 1.0f - wy1;
  int x0 = (int)x0f, y0 = (int)y0f;
  int x1 = x0 + 1,  y1 = y0 + 1;
  bool bx0 = (x0 >= 0) & (x0 < HW), bx1 = (x1 >= 0) & (x1 < HW);
  bool by0 = (y0 >= 0) & (y0 < HW), by1 = (y1 >= 0) & (y1 < HW);
  float v00 = (bx0 & by0) ? fmap[y0 * HW + x0] : 0.0f;
  float v10 = (bx1 & by0) ? fmap[y0 * HW + x1] : 0.0f;
  float v01 = (bx0 & by1) ? fmap[y1 * HW + x0] : 0.0f;
  float v11 = (bx1 & by1) ? fmap[y1 * HW + x1] : 0.0f;
  return v00 * wx0 * wy0 + v10 * wx1 * wy0 + v01 * wx0 * wy1 + v11 * wx1 * wy1;
}

// ---------------- weight pre-swizzle (f32 -> bf16, WMMA B-fragment order) ----------------
// B fragment element mapping: N = tN*16 + (lane&15), K = tK*32 + (elem&8)*2 + (lane>>4)*8 + (elem&7)
__global__ void swizzle_weights(const float* __restrict__ w_u_w,
                                const float* __restrict__ w_delta_w,
                                const float* __restrict__ w_a_w,
                                const float* __restrict__ w_o_w,
                                bf16* __restrict__ ws) {
  int e = blockIdx.x * blockDim.x + threadIdx.x;
  if (e >= WS_TOTAL) return;
  float v;
  if (e < WU_ELEMS) {
    int el = e, frag = el >> 9, rem = el & 511, ln = rem >> 4, elem = rem & 15;
    int tN = frag / WU_KT, tK = frag - tN * WU_KT;
    int N = tN * 16 + (ln & 15);
    int Kv = tK * 32 + (elem & 8) * 2 + (ln >> 4) * 8 + (elem & 7);
    v = w_u_w[N * UIN + Kv];
  } else if (e < WS_WO_OFF) {
    int el = e - WS_WDA_OFF, frag = el >> 9, rem = el & 511, ln = rem >> 4, elem = rem & 15;
    int tN = frag / WDA_KT, tK = frag - tN * WDA_KT;
    int N = tN * 16 + (ln & 15);
    int Kv = tK * 32 + (elem & 8) * 2 + (ln >> 4) * 8 + (elem & 7);
    v = (N < 144) ? w_delta_w[N * D_MODEL + Kv]
                  : (N < 216 ? w_a_w[(N - 144) * D_MODEL + Kv] : 0.0f);
  } else {
    int el = e - WS_WO_OFF, frag = el >> 9, rem = el & 511, ln = rem >> 4, elem = rem & 15;
    int tN = frag / WO_KT, tK = frag - tN * WO_KT;
    int N = tN * 16 + (ln & 15);
    int Kv = tK * 32 + (elem & 8) * 2 + (ln >> 4) * 8 + (elem & 7);
    v = w_o_w[N * D_MODEL + Kv];
  }
  ws[e] = (bf16)v;
}

// ---------------- fused main kernel: one wave = one 16-item tile ----------------
struct __align__(16) WaveLds {
  union {
    bf16  a_bf[16 * UIN];      // u_input tile  (16 x 416) bf16, GEMM1 A
    float g2[16 * G2COLS];     // GEMM2 output  (16 x 224) f32 (offsets+logits)
  } sA;
  bf16 sU[16 * D_MODEL];       // LN(GELU(u)) tile, GEMM2/3 A
  bf16 sHr[16 * D_MODEL];      // sampled h_r tile, GEMM4 A
  float anc[16][2];            // anchor_px per row
};

__global__ __launch_bounds__(64)
void deform_main(const float* __restrict__ h, const int* __restrict__ top_indices,
                 const float* __restrict__ qc, const float* __restrict__ g,
                 const float* __restrict__ L2p, const float* __restrict__ L3p,
                 const float* __restrict__ L4p,
                 const float* __restrict__ w_u_b, const float* __restrict__ ln_g,
                 const float* __restrict__ ln_b, const float* __restrict__ w_delta_b,
                 const float* __restrict__ w_a_b, const float* __restrict__ w_o_b,
                 const float* __restrict__ e_def,
                 const bf16* __restrict__ ws, float* __restrict__ out) {
  __shared__ WaveLds lds[2];
  const int lane = threadIdx.x & 31;
  const int wv   = threadIdx.x >> 5;
  WaveLds& W = lds[wv];

  const int tile = blockIdx.x * 2 + wv;
  const int i0 = tile * 16;                 // 16 consecutive r's share (b,k)
  const int b  = i0 / (KK * RR);
  const int rm = i0 - b * KK * RR;
  const int k  = rm / RR;
  const int r0 = rm - k * RR;

  const int l15 = lane & 15;
  const int lh  = lane >> 4;

  // ---- stage u_input = [h | g_r | fourier_pe] as bf16 into LDS ----
  {
    const int srow = l15;                   // each lane pair (lh=0/1) covers one row
    int ti = top_indices[(b * KK + k) * RR + r0 + srow];
    float ax = (float)(ti & 31) * 32.0f + 16.0f;
    float ay = (float)(ti >> 5) * 32.0f + 16.0f;
    if (lh == 0) { W.anc[srow][0] = ax; W.anc[srow][1] = ay; }
    float qx = qc[(b * KK + k) * 2 + 0];
    float qy = qc[(b * KK + k) * 2 + 1];
    float dnx = (ax - qx) * (1.0f / 1024.0f);
    float dny = (ay - qy) * (1.0f / 1024.0f);
    const float* hrow = h + (b * KK + k) * D_MODEL;
    const float* grow = g + (b * 1024 + ti) * D_MODEL;
    for (int t = 0; t < 208; ++t) {
      int c = lh * 208 + t;
      float v;
      if (c < 192)      v = hrow[c];
      else if (c < 384) v = grow[c - 192];
      else {
        int j = c - 384;
        float f = (float)(1 << (j & 7));
        float ang = 6.28318530717958647f * f * ((j & 16) ? dny : dnx);
        v = (j & 8) ? cosf(ang) : sinf(ang);
      }
      W.sA.a_bf[srow * UIN + c] = (bf16)v;
    }
  }

  // ---- GEMM1: u = GELU(u_input @ W_u^T + b_u), 16x416 * 416x192 ----
  const bf16* wsWu = ws;
  for (int tN = 0; tN < WU_NT; ++tN) {
    v8f acc = {0.f, 0.f, 0.f, 0.f, 0.f, 0.f, 0.f, 0.f};
#pragma unroll
    for (int tK = 0; tK < WU_KT; ++tK) {
      v16bf a = load_a_frag(W.sA.a_bf, UIN, tK * 32, lane);
      v16bf bf = load_b_frag(wsWu + (tN * WU_KT + tK) * FRAG_ELEMS, lane);
      acc = wmma_bf16(a, bf, acc);
    }
    int n = tN * 16 + l15;
    float bias = w_u_b[n];
#pragma unroll
    for (int j = 0; j < 8; ++j) {
      int r = j + lh * 8;
      float x = acc[j] + bias;
      float ge = 0.5f * x * (1.0f + erff(x * 0.70710678118654752f)); // exact GELU
      W.sU[r * D_MODEL + n] = (bf16)ge;
    }
  }

  // ---- LayerNorm over 192 per row, in place on sU (bf16) ----
  {
    int r = l15;
    float s = 0.f, s2 = 0.f;
    for (int c0 = 0; c0 < 96; ++c0) {
      int c = lh * 96 + c0;
      float v = (float)W.sU[r * D_MODEL + c];
      s += v; s2 += v * v;
    }
    s  += __shfl_xor(s, 16, 32);
    s2 += __shfl_xor(s2, 16, 32);
    float mu  = s * (1.0f / 192.0f);
    float var = s2 * (1.0f / 192.0f) - mu * mu;
    float inv = rsqrtf(var + 1e-5f);
    for (int c0 = 0; c0 < 96; ++c0) {
      int c = lh * 96 + c0;
      float v = (float)W.sU[r * D_MODEL + c];
      v = (v - mu) * inv * ln_g[c] + ln_b[c];
      W.sU[r * D_MODEL + c] = (bf16)v;
    }
  }

  // ---- GEMM2: [offsets(144) | logits(72) | pad] = u @ W_da^T, 16x192 * 192x224 ----
  const bf16* wsDa = ws + WS_WDA_OFF;
  for (int tN = 0; tN < WDA_NT; ++tN) {
    v8f acc = {0.f, 0.f, 0.f, 0.f, 0.f, 0.f, 0.f, 0.f};
#pragma unroll
    for (int tK = 0; tK < WDA_KT; ++tK) {
      v16bf a = load_a_frag(W.sU, D_MODEL, tK * 32, lane);
      v16bf bf = load_b_frag(wsDa + (tN * WDA_KT + tK) * FRAG_ELEMS, lane);
      acc = wmma_bf16(a, bf, acc);
    }
    int n = tN * 16 + l15;
    float bias = (n < 144) ? w_delta_b[n] : (n < 216 ? w_a_b[n - 144] : 0.0f);
#pragma unroll
    for (int j = 0; j < 8; ++j) {
      int r = j + lh * 8;
      W.sA.g2[r * G2COLS + n] = acc[j] + bias;   // reuse sA region (u_input dead)
    }
  }

  // ---- softmax + tanh-offset bilinear sampling; lane = channel within head ----
  {
    const float* projs[3] = {L2p, L3p, L4p};
    const int   hw[3]   = {128, 64, 32};
    const float strd[3] = {8.f, 16.f, 32.f};
    const float sigm[3] = {4.f, 2.f, 1.f};
    for (int it = 0; it < 16 * NHEAD; ++it) {
      int item = it / NHEAD;
      int head = it - item * NHEAD;
      const float* rowp = &W.sA.g2[item * G2COLS];
      float lg[12];
      float mx = -1e30f;
#pragma unroll
      for (int j = 0; j < 12; ++j) { lg[j] = rowp[144 + head * 12 + j]; mx = fmaxf(mx, lg[j]); }
      float ssum = 0.f;
#pragma unroll
      for (int j = 0; j < 12; ++j) { lg[j] = expf(lg[j] - mx); ssum += lg[j]; }
      float rinv = 1.0f / ssum;
      float axc = W.anc[item][0], ayc = W.anc[item][1];
      int chan = head * DH + lane;
      float acc = 0.f;
#pragma unroll
      for (int l = 0; l < 3; ++l) {
        int HW = hw[l];
        const float* fmap = projs[l] + (size_t)(b * D_MODEL + chan) * HW * HW;
        float bax = axc / strd[l], bay = ayc / strd[l];
#pragma unroll
        for (int m = 0; m < 4; ++m) {
          float ox = rowp[head * 24 + l * 8 + m * 2 + 0];
          float oy = rowp[head * 24 + l * 8 + m * 2 + 1];
          float px = bax + tanhf(ox) * sigm[l];
          float py = bay + tanhf(oy) * sigm[l];
          acc += (lg[l * 4 + m] * rinv) * bilin(fmap, HW, px, py);
        }
      }
      W.sHr[item * D_MODEL + chan] = (bf16)acc;
    }
  }

  // ---- GEMM4: out = h_r @ W_o^T + b_o + e_deform, 16x192 * 192x192 -> global ----
  const bf16* wsO = ws + WS_WO_OFF;
  size_t outbase = ((size_t)(b * KK + k) * RR + r0) * D_MODEL;
  for (int tN = 0; tN < WO_NT; ++tN) {
    v8f acc = {0.f, 0.f, 0.f, 0.f, 0.f, 0.f, 0.f, 0.f};
#pragma unroll
    for (int tK = 0; tK < WO_KT; ++tK) {
      v16bf a = load_a_frag(W.sHr, D_MODEL, tK * 32, lane);
      v16bf bf = load_b_frag(wsO + (tN * WO_KT + tK) * FRAG_ELEMS, lane);
      acc = wmma_bf16(a, bf, acc);
    }
    int n = tN * 16 + l15;
    float bo = w_o_b[n] + e_def[n];
#pragma unroll
    for (int j = 0; j < 8; ++j) {
      int r = j + lh * 8;
      out[outbase + (size_t)r * D_MODEL + n] = acc[j] + bo;
    }
  }
}

// ---------------- host entry ----------------
extern "C" void kernel_launch(void* const* d_in, const int* in_sizes, int n_in,
                              void* d_out, int out_size, void* d_ws, size_t ws_size,
                              hipStream_t stream) {
  (void)in_sizes; (void)n_in; (void)out_size; (void)ws_size;
  const float* h        = (const float*)d_in[0];
  const int*   ti       = (const int*)d_in[1];
  const float* qc       = (const float*)d_in[2];
  const float* g        = (const float*)d_in[3];
  const float* L2p      = (const float*)d_in[4];
  const float* L3p      = (const float*)d_in[5];
  const float* L4p      = (const float*)d_in[6];
  const float* w_u_w    = (const float*)d_in[7];
  const float* w_u_b    = (const float*)d_in[8];
  const float* ln_g     = (const float*)d_in[9];
  const float* ln_b     = (const float*)d_in[10];
  const float* w_delta_w= (const float*)d_in[11];
  const float* w_delta_b= (const float*)d_in[12];
  const float* w_a_w    = (const float*)d_in[13];
  const float* w_a_b    = (const float*)d_in[14];
  const float* w_o_w    = (const float*)d_in[15];
  const float* w_o_b    = (const float*)d_in[16];
  const float* e_def    = (const float*)d_in[17];
  bf16* ws = (bf16*)d_ws;
  float* out = (float*)d_out;

  swizzle_weights<<<(WS_TOTAL + 255) / 256, 256, 0, stream>>>(
      w_u_w, w_delta_w, w_a_w, w_o_w, ws);

  // 16384 items / 16 per wave / 2 waves per block = 512 blocks of 64 threads
  deform_main<<<512, 64, 0, stream>>>(h, ti, qc, g, L2p, L3p, L4p,
                                      w_u_b, ln_g, ln_b, w_delta_b, w_a_b,
                                      w_o_b, e_def, ws, out);
}
